// BaseAttention_22651657519434
// MI455X (gfx1250) — matrix operand order repaired and checked by reading, not compiled, not run on hardware
//
#include <hip/hip_runtime.h>
#include <hip/hip_bf16.h>

// ---------------------------------------------------------------------------
// MI455X (gfx1250) attention block:
//   pre : x, w_attn, w_proj -> bf16
//   qkv = x @ w_attn^T            (WMMA bf16 GEMM, async-LDS double buffered)
//   q,k = rmsnorm(rope(q,k)); v   (VALU; bf16 Q/K [B,H,T,D], V^T [B,H,D,T])
//   y   = causal softmax(q k^T / sqrt(D)) v   (flash, WMMA bf16) -> bf16
//   out = y @ w_proj^T            (WMMA bf16 GEMM, async-LDS double buffered)
// ---------------------------------------------------------------------------

typedef __attribute__((ext_vector_type(16))) __bf16 v16bf;
typedef __attribute__((ext_vector_type(8)))  __bf16 v8bf;
typedef __attribute__((ext_vector_type(8)))  float  v8f;

#define N_B   4
#define N_T   2048
#define N_C   1024
#define N_H   16
#define N_D   64

__device__ __forceinline__ v8f wmma_bf16(v16bf a, v16bf b, v8f c) {
    return __builtin_amdgcn_wmma_f32_16x16x32_bf16(
        false, a, false, b, (short)0, c, false, false);
}

// 16-byte async copy global -> LDS (ASYNCcnt-tracked, per-lane addresses).
// Generic pointers to LDS carry the LDS byte offset in addr[31:0].
__device__ __forceinline__ void async_cp16(__bf16* ldst, const __bf16* gsrc) {
    unsigned l = (unsigned)(unsigned long long)ldst;
    asm volatile("global_load_async_to_lds_b128 %0, %1, off"
                 :: "v"(l), "v"(gsrc) : "memory");
}

// A-fragment (16x32 bf16): lane m=lane&15; elems 0..7 = K c0..c0+7,
// elems 8..15 = K c0+16..c0+23, c0 = (lane>=16)*8. hi half at p+16.
__device__ __forceinline__ v16bf ldfragA(const __bf16* p) {
    v8bf lo = *(const v8bf*)(p);
    v8bf hi = *(const v8bf*)(p + 16);
    return __builtin_shufflevector(lo, hi, 0,1,2,3,4,5,6,7,8,9,10,11,12,13,14,15);
}

// B-fragment (32x16 bf16): lane n=lane&15; elems 0..7 = K kb..kb+7,
// elems 8..15 = K kb+16..kb+23, kb = (lane>=16)*8. hi half at p+16.
__device__ __forceinline__ v16bf ldfragB(const __bf16* p) {
    v8bf lo = *(const v8bf*)(p);
    v8bf hi = *(const v8bf*)(p + 16);
    return __builtin_shufflevector(lo, hi, 0,1,2,3,4,5,6,7,8,9,10,11,12,13,14,15);
}

__device__ __forceinline__ float red16_max(float v) {
    #pragma unroll
    for (int m = 1; m < 16; m <<= 1) v = fmaxf(v, __shfl_xor(v, m, 32));
    return v;
}
__device__ __forceinline__ float red16_sum(float v) {
    #pragma unroll
    for (int m = 1; m < 16; m <<= 1) v += __shfl_xor(v, m, 32);
    return v;
}
__device__ __forceinline__ float red32_sum(float v) {
    #pragma unroll
    for (int m = 1; m < 32; m <<= 1) v += __shfl_xor(v, m, 32);
    return v;
}

// ---------------------------------------------------------------------------
// f32 -> bf16 conversion pass (8 elements / thread, 16B stores)
// ---------------------------------------------------------------------------
__global__ __launch_bounds__(256) void cvt_f32_bf16(
    const float* __restrict__ src, __bf16* __restrict__ dst, int n8)
{
    int i = blockIdx.x * 256 + threadIdx.x;
    if (i >= n8) return;
    float4 a = ((const float4*)src)[i * 2 + 0];
    float4 b = ((const float4*)src)[i * 2 + 1];
    v8bf o;
    o[0] = (__bf16)a.x; o[1] = (__bf16)a.y; o[2] = (__bf16)a.z; o[3] = (__bf16)a.w;
    o[4] = (__bf16)b.x; o[5] = (__bf16)b.y; o[6] = (__bf16)b.z; o[7] = (__bf16)b.w;
    *(v8bf*)(dst + (size_t)i * 8) = o;
}

// ---------------------------------------------------------------------------
// C[M,N](f32) = A[M,K](bf16) @ B[N,K](bf16)^T. Block tile 64x128, BK=64,
// 256 threads = 8 waves. Double-buffered LDS fed by async global->LDS copies:
// tile k+1 streams in while tile k is consumed by 8 back-to-back WMMAs/wave.
// ---------------------------------------------------------------------------
__global__ __launch_bounds__(256) void gemm_nt_async(
    const __bf16* __restrict__ A, const __bf16* __restrict__ Bw,
    float* __restrict__ C, int lda, int ldb, int ldc, int K)
{
    __shared__ __bf16 As[2][64][72];     // +8 pad; rows stay 16B aligned
    __shared__ __bf16 Bs[2][128][72];

    const int tid  = threadIdx.x;
    const int lane = tid & 31;
    const int wave = tid >> 5;
    const int wm   = wave & 3;           // 4 row tiles of 16
    const int wn   = wave >> 2;          // 2 col halves of 64
    const int n16  = lane & 15;
    const int half = lane >> 4;
    const int c0   = half * 8;
    const int kb   = half * 8;

    const int m0 = blockIdx.y * 64;
    const int n0 = blockIdx.x * 128;

    v8f acc[4] = {};

    // issue 6 async 16B copies per thread for one (A,B) tile pair
    auto stage = [&](int buf, int kt) {
        #pragma unroll
        for (int it = 0; it < 2; ++it) {            // A: 64x64 bf16 = 512 chunks
            int idx = tid + it * 256;
            int r = idx >> 3, c = (idx & 7) * 8;
            async_cp16(&As[buf][r][c], A + (size_t)(m0 + r) * lda + kt + c);
        }
        #pragma unroll
        for (int it = 0; it < 4; ++it) {            // B: 128x64 bf16 = 1024 chunks
            int idx = tid + it * 256;
            int r = idx >> 3, c = (idx & 7) * 8;
            async_cp16(&Bs[buf][r][c], Bw + (size_t)(n0 + r) * ldb + kt + c);
        }
    };

    stage(0, 0);

    const int NT = K / 64;
    for (int ki = 0; ki < NT; ++ki) {
        const int buf = ki & 1;
        if (ki + 1 < NT) {
            stage(buf ^ 1, (ki + 1) * 64);          // prefetch next tile
            asm volatile("s_wait_asynccnt 0x6" ::: "memory");  // cur tile done
        } else {
            asm volatile("s_wait_asynccnt 0x0" ::: "memory");
        }
        __syncthreads();

        // hoist all fragments, then 8 hazard-free WMMAs
        v16bf aA0 = ldfragA(&As[buf][wm * 16 + n16][c0]);
        v16bf aA1 = ldfragA(&As[buf][wm * 16 + n16][32 + c0]);
        v16bf bB[2][4];
        #pragma unroll
        for (int kk = 0; kk < 2; ++kk)
            #pragma unroll
            for (int dt = 0; dt < 4; ++dt)
                bB[kk][dt] = ldfragB(&Bs[buf][wn * 64 + dt * 16 + n16][kk * 32 + kb]);
        #pragma unroll
        for (int dt = 0; dt < 4; ++dt) {
            acc[dt] = wmma_bf16(aA0, bB[0][dt], acc[dt]);
            acc[dt] = wmma_bf16(aA1, bB[1][dt], acc[dt]);
        }
        __syncthreads();                            // done reading buf
    }

    #pragma unroll
    for (int dt = 0; dt < 4; ++dt)
        #pragma unroll
        for (int i = 0; i < 8; ++i)
            C[(size_t)(m0 + wm * 16 + i + half * 8) * ldc
              + n0 + wn * 64 + dt * 16 + n16] = acc[dt][i];
}

// ---------------------------------------------------------------------------
// RoPE + RMSNorm on q,k; emit bf16 q,k [B,H,T,D] and bf16 v^T [B,H,D,T].
// One wave per (token, head); lane indexes d_half=32.
// ---------------------------------------------------------------------------
__global__ __launch_bounds__(256) void rope_rms_cvt(
    const float* __restrict__ qkv, const float* __restrict__ cosp,
    const float* __restrict__ sinp, __bf16* __restrict__ qb,
    __bf16* __restrict__ kb, __bf16* __restrict__ vt)
{
    const int lane = threadIdx.x & 31;
    const int wave = threadIdx.x >> 5;
    const long task = (long)blockIdx.x * 8 + wave;   // 0 .. B*T*H-1
    const int  h = (int)(task & (N_H - 1));
    const long m = task >> 4;                        // 0 .. B*T-1
    const int  t = (int)(m & (N_T - 1));
    const int  b = (int)(m >> 11);

    const float c = cosp[t * 32 + lane];
    const float s = sinp[t * 32 + lane];
    const float* base = qkv + m * (3 * N_C) + h * N_D;
    const size_t bh = (size_t)(b * N_H + h);

    {   // q
        float x1 = base[lane], x2 = base[32 + lane];
        float r1 =  x1 * c + x2 * s;
        float r2 = -x1 * s + x2 * c;
        float ss = red32_sum(r1 * r1 + r2 * r2);
        float inv = rsqrtf(ss * (1.0f / N_D) + 1e-6f);
        size_t o = (bh * N_T + t) * N_D;
        qb[o + lane]      = (__bf16)(r1 * inv);
        qb[o + 32 + lane] = (__bf16)(r2 * inv);
    }
    {   // k
        float x1 = base[N_C + lane], x2 = base[N_C + 32 + lane];
        float r1 =  x1 * c + x2 * s;
        float r2 = -x1 * s + x2 * c;
        float ss = red32_sum(r1 * r1 + r2 * r2);
        float inv = rsqrtf(ss * (1.0f / N_D) + 1e-6f);
        size_t o = (bh * N_T + t) * N_D;
        kb[o + lane]      = (__bf16)(r1 * inv);
        kb[o + 32 + lane] = (__bf16)(r2 * inv);
    }
    {   // v -> transposed vt[bh][d][t]
        float v1 = base[2 * N_C + lane];
        float v2 = base[2 * N_C + 32 + lane];
        size_t o = bh * N_D * N_T;
        vt[o + (size_t)lane * N_T + t]        = (__bf16)v1;
        vt[o + (size_t)(32 + lane) * N_T + t] = (__bf16)v2;
    }
}

// ---------------------------------------------------------------------------
// Causal flash attention. One wave per 16-row Q tile per (b,h); key blocks
// of 32; scores 4 wmma, P.V 4 wmma; P re-laned C->A layout through LDS.
// Emits y in bf16 [B,T,C] for the projection GEMM.
// ---------------------------------------------------------------------------
__global__ __launch_bounds__(128) void flash_attn(
    const __bf16* __restrict__ qb, const __bf16* __restrict__ kbm,
    const __bf16* __restrict__ vt, __bf16* __restrict__ yb)
{
    __shared__ __bf16 lds_p[4][16 * 32];             // per-wave P staging

    const int lane = threadIdx.x & 31;
    const int wave = threadIdx.x >> 5;
    const int task = blockIdx.x * 4 + wave;          // 0 .. B*H*(T/16)-1
    const int mt   = task & (N_T / 16 - 1);
    const int bh   = task >> 7;
    const int b    = bh >> 4, h = bh & 15;
    const int mbase = mt * 16;

    const int n16  = lane & 15;
    const int half = lane >> 4;
    const int c0   = half * 8;
    const int kb16 = half * 8;

    const __bf16* qbase = qb  + (size_t)bh * N_T * N_D;
    const __bf16* kbase = kbm + (size_t)bh * N_T * N_D;
    const __bf16* vbase = vt  + (size_t)bh * N_D * N_T;
    __bf16* P = lds_p[wave];

    const __bf16* qrow = qbase + (size_t)(mbase + n16) * N_D;
    v16bf aQ0 = ldfragA(qrow + c0);          // head dims 0..31
    v16bf aQ1 = ldfragA(qrow + 32 + c0);     // head dims 32..63

    v8f   o[4] = {};
    float mrow[8], lrow[8];
    #pragma unroll
    for (int i = 0; i < 8; ++i) { mrow[i] = -3.0e38f; lrow[i] = 0.0f; }

    const float scale = 0.125f;              // 1/sqrt(64)

    for (int jb = 0; jb < mbase + 16; jb += 32) {
        v8f s[2] = {};
        #pragma unroll
        for (int ct = 0; ct < 2; ++ct) {
            const __bf16* krow = kbase + (size_t)(jb + ct * 16 + n16) * N_D;
            v16bf bK0 = ldfragB(krow + kb16);
            v16bf bK1 = ldfragB(krow + 32 + kb16);
            s[ct] = wmma_bf16(aQ0, bK0, s[ct]);
            s[ct] = wmma_bf16(aQ1, bK1, s[ct]);
        }

        #pragma unroll
        for (int i = 0; i < 8; ++i) {
            const int row_g = mbase + i + half * 8;
            float s0 = s[0][i] * scale;
            float s1 = s[1][i] * scale;
            if (jb + n16      > row_g) s0 = -3.0e38f;
            if (jb + 16 + n16 > row_g) s1 = -3.0e38f;

            float mx = red16_max(fmaxf(s0, s1));
            float mn = fmaxf(mrow[i], mx);
            float alpha = __expf(mrow[i] - mn);
            float p0 = __expf(s0 - mn);
            float p1 = __expf(s1 - mn);
            float rs = red16_sum(p0 + p1);
            lrow[i] = lrow[i] * alpha + rs;
            mrow[i] = mn;
            #pragma unroll
            for (int dt = 0; dt < 4; ++dt) o[dt][i] *= alpha;

            P[(i + half * 8) * 32 + n16]      = (__bf16)p0;
            P[(i + half * 8) * 32 + 16 + n16] = (__bf16)p1;
        }

        asm volatile("s_wait_dscnt 0" ::: "memory");   // cross-lane LDS RAW

        v16bf aP = ldfragA(&P[n16 * 32 + c0]);
        #pragma unroll
        for (int dt = 0; dt < 4; ++dt) {
            v16bf bV = ldfragB(vbase + (size_t)(dt * 16 + n16) * N_T + jb + kb16);
            o[dt] = wmma_bf16(aP, bV, o[dt]);
        }
        asm volatile("s_wait_dscnt 0" ::: "memory");   // before P overwrite
    }

    #pragma unroll
    for (int i = 0; i < 8; ++i) {
        const float inv = 1.0f / lrow[i];
        const size_t yo = (size_t)(b * N_T + mbase + i + half * 8) * N_C + h * N_D;
        #pragma unroll
        for (int dt = 0; dt < 4; ++dt)
            yb[yo + dt * 16 + n16] = (__bf16)(o[dt][i] * inv);
    }
}

// ---------------------------------------------------------------------------
extern "C" void kernel_launch(void* const* d_in, const int* in_sizes, int n_in,
                              void* d_out, int out_size, void* d_ws, size_t ws_size,
                              hipStream_t stream) {
    const float* x    = (const float*)d_in[0];
    const float* cosp = (const float*)d_in[1];
    const float* sinp = (const float*)d_in[2];
    const float* wa   = (const float*)d_in[3];
    const float* wp   = (const float*)d_in[4];
    float* out = (float*)d_out;

    const size_t MT = (size_t)N_B * N_T;                 // 8192 tokens
    char* ws = (char*)d_ws;
    float*  qkv = (float*)ws;                            // 96 MB
    char* p = ws + MT * 3 * N_C * sizeof(float);
    __bf16* xb  = (__bf16*)p;  p += MT * N_C * 2;        // x   bf16
    __bf16* wab = (__bf16*)p;  p += (size_t)3 * N_C * N_C * 2;
    __bf16* wpb = (__bf16*)p;  p += (size_t)N_C * N_C * 2;
    __bf16* qb  = (__bf16*)p;  p += MT * N_C * 2;
    __bf16* kb  = (__bf16*)p;  p += MT * N_C * 2;
    __bf16* vt  = (__bf16*)p;  p += MT * N_C * 2;
    __bf16* yb  = (__bf16*)p;

    // 0) downconvert operands to bf16
    {
        int n8;
        n8 = (int)(MT * N_C / 8);
        cvt_f32_bf16<<<(n8 + 255) / 256, 256, 0, stream>>>(x, xb, n8);
        n8 = 3 * N_C * N_C / 8;
        cvt_f32_bf16<<<(n8 + 255) / 256, 256, 0, stream>>>(wa, wab, n8);
        n8 = N_C * N_C / 8;
        cvt_f32_bf16<<<(n8 + 255) / 256, 256, 0, stream>>>(wp, wpb, n8);
    }

    // 1) qkv = x @ w_attn^T   (8192 x 3072 x 1024)
    gemm_nt_async<<<dim3(3 * N_C / 128, MT / 64), 256, 0, stream>>>(
        xb, wab, qkv, N_C, N_C, 3 * N_C, N_C);

    // 2) RoPE + RMSNorm + bf16 conversion / V transpose
    rope_rms_cvt<<<(N_B * N_T * N_H) / 8, 256, 0, stream>>>(
        qkv, cosp, sinp, qb, kb, vt);

    // 3) causal flash attention -> y bf16 [B,T,C]
    flash_attn<<<(N_B * N_H * (N_T / 16)) / 4, 128, 0, stream>>>(qb, kb, vt, yb);

    // 4) out = y @ w_proj^T   (8192 x 1024 x 1024)
    gemm_nt_async<<<dim3(N_C / 128, MT / 64), 256, 0, stream>>>(
        yb, wpb, out, N_C, N_C, N_C, N_C);
}
